// grafiti__25391846654127
// MI455X (gfx1250) — matrix operand order repaired and checked by reading, not compiled, not run on hardware
//
#include <hip/hip_runtime.h>
#include <hip/hip_bf16.h>
#include <math.h>

// ---------------------------------------------------------------------------
// GraFITi forward for MI455X (gfx1250, wave32, WMMA f32_16x16x32_f16)
// ---------------------------------------------------------------------------

typedef _Float16 v16h __attribute__((ext_vector_type(16)));
typedef float    v8f  __attribute__((ext_vector_type(8)));

constexpr int cB = 8, cT = 128, cD = 41, cM = 128, cH = 4, cL = 3;
constexpr int cN  = cT * cD;      // 5248 positions per batch
constexpr int cBn = cB * cN;      // 41984 total rows

// ---------------------------------------------------------------------------
// WMMA GEMM: Y(rows x 128) = act(X(rows x K) @ W(128 x K)^T + bias)
// Compile-time epilogue: RELU, RESID (fp32 residual add), SCALE (per-row
// scale before residual), YBF (also emit f16 copy).
// Block = 256 threads = 8 waves; one block owns a 16-row x 128-col panel.
// The 16 x K f16 A panel is staged once in LDS (row-padded by 8 halfs to
// spread rows across LDS banks); each wave computes one 16x16 column tile.
// Out-of-range A rows are clamped (never stored), so the K-loop has no
// divergence and EXEC stays all-ones for every v_wmma.  The store epilogue
// branches ONCE (wave-uniform) on full-tile vs partial-tile, so the common
// path is straight-line stores with no EXEC churn.
// ---------------------------------------------------------------------------
template <int K, bool RELU, bool RESID, bool SCALE, bool YBF>
__global__ __launch_bounds__(256)
void gemm_f16_wmma(const _Float16* __restrict__ X, const _Float16* __restrict__ W,
                   const float* __restrict__ bias, const float* __restrict__ resid,
                   const float* __restrict__ rowscale,
                   float* __restrict__ Y, _Float16* __restrict__ Ybf,
                   int rows)
{
    constexpr int KP = K + 8;                 // padded row stride (halfs)
    __shared__ _Float16 As[16 * KP];

    const int rowBase = blockIdx.x * 16;

    // ---- cooperative A-panel load: global (clamped rows) -> LDS ----
    {
        const uint32_t* Xw = reinterpret_cast<const uint32_t*>(X);
        uint32_t* Aw = reinterpret_cast<uint32_t*>(As);
        constexpr int rowDw = K / 2;          // dwords per source row
        constexpr int totalDw = 16 * rowDw;
#pragma unroll
        for (int idx = 0; idx < totalDw; idx += 256) {
            const int t = idx + threadIdx.x;
            const int r = t / rowDw;
            const int c = t - r * rowDw;
            int gr = rowBase + r;
            if (gr >= rows) gr = rows - 1;    // clamp: compute garbage, never store
            Aw[r * (KP / 2) + c] = Xw[(size_t)gr * rowDw + c];
        }
    }
    __syncthreads();

    const int wave = threadIdx.x >> 5;
    const int lane = threadIdx.x & 31;
    const int mrow = lane & 15;
    const int hi   = lane >> 4;
    const int ncol = wave * 16 + (lane & 15); // column this lane owns (0..127)

    v8f acc = {};
#pragma unroll
    for (int k0 = 0; k0 < K; k0 += 32) {
        // ---- A tile (16x32 f16) from LDS: two contiguous 8-half chunks ----
        v16h a;
        const _Float16* ar0 = &As[mrow * KP + k0 + 16 * hi];
        const _Float16* ar1 = &As[mrow * KP + k0 + 16 * hi + 8];
#pragma unroll
        for (int e = 0; e < 8; ++e) a[e] = ar0[e];
#pragma unroll
        for (int e = 0; e < 8; ++e) a[8 + e] = ar1[e];

        // ---- B tile (32x16 f16) = W^T: 16 contiguous K per half-wave ----
        v16h bm;
        const _Float16* wr = W + (size_t)ncol * K + k0 + 16 * hi;
#pragma unroll
        for (int e = 0; e < 16; ++e) bm[e] = wr[e];

        acc = __builtin_amdgcn_wmma_f32_16x16x32_f16(
            false, a, false, bm, (short)0, acc, false, false);
    }

    const float bv = bias[ncol];
    auto emit = [&](int r) {
        const int orow = rowBase + r + 8 * hi;
        float v = acc[r] + bv;
        if (RELU)  v = fmaxf(v, 0.0f);
        if (SCALE) v *= rowscale[orow];
        const size_t off = (size_t)orow * 128 + ncol;
        if (RESID) v += resid[off];
        Y[off] = v;
        if (YBF) Ybf[off] = (_Float16)v;
    };

    if (rowBase + 16 <= rows) {               // wave-uniform: full tile
#pragma unroll
        for (int r = 0; r < 8; ++r) emit(r);
    } else {                                  // partial tile (rare, tiny GEMMs)
#pragma unroll
        for (int r = 0; r < 8; ++r)
            if (rowBase + r + 8 * hi < rows) emit(r);
    }
}

// ---------------------------------------------------------------------------
// Masked online-softmax attention.  One block per (query row, head, batch).
// scores = <q_h, k_h>/sqrt(M) where ind[key]==qi && mk>0, else -1e10.
// O = q_h + softmax(scores) @ V_h   (exactly matches reference semantics,
// including the all-masked -> uniform case).
// ---------------------------------------------------------------------------
__global__ __launch_bounds__(256)
void attn_kernel(const float* __restrict__ qp, const float* __restrict__ kp,
                 const float* __restrict__ vp, const int* __restrict__ ind,
                 const float* __restrict__ mk,
                 float* __restrict__ O, _Float16* __restrict__ Obf,
                 int NQ, int Nk, float scale)
{
    const int qi = blockIdx.x, h = blockIdx.y, b = blockIdx.z;
    const int tid = threadIdx.x;

    __shared__ float qs[32];
    __shared__ float sm[256];
    __shared__ float ss[256];
    __shared__ float sacc[256 * 32];

    const float* qrow = qp + ((size_t)(b * NQ + qi) * 128) + h * 32;
    if (tid < 32) qs[tid] = qrow[tid];
    __syncthreads();

    float m = -1e30f, s = 0.0f;
    float acc[32];
#pragma unroll
    for (int d = 0; d < 32; ++d) acc[d] = 0.0f;

    const int base = b * Nk;
    for (int ki = tid; ki < Nk; ki += 256) {
        const bool on = (ind[base + ki] == qi) && (mk[base + ki] > 0.0f);
        float sc;
        if (on) {
            const float* kr = kp + ((size_t)(base + ki) * 128) + h * 32;
            float dot = 0.0f;
#pragma unroll
            for (int d = 0; d < 32; ++d) dot += qs[d] * kr[d];
            sc = dot * scale;
        } else {
            sc = -1e10f;
        }
        const float mn = fmaxf(m, sc);
        const float ea = __expf(m - mn);
        const float eb = __expf(sc - mn);
        s = s * ea + eb;
#pragma unroll
        for (int d = 0; d < 32; ++d) acc[d] *= ea;
        if (eb > 0.0f) {
            const float* vr = vp + ((size_t)(base + ki) * 128) + h * 32;
#pragma unroll
            for (int d = 0; d < 32; ++d) acc[d] += eb * vr[d];
        }
        m = mn;
    }

    sm[tid] = m; ss[tid] = s;
    for (int d = 0; d < 32; ++d) sacc[tid * 32 + d] = acc[d];
    __syncthreads();

    for (int st = 128; st > 0; st >>= 1) {
        if (tid < st) {
            const float m1 = sm[tid], m2 = sm[tid + st];
            const float mn = fmaxf(m1, m2);
            const float e1 = __expf(m1 - mn), e2 = __expf(m2 - mn);
            sm[tid] = mn;
            ss[tid] = ss[tid] * e1 + ss[tid + st] * e2;
            for (int d = 0; d < 32; ++d)
                sacc[tid * 32 + d] = sacc[tid * 32 + d] * e1 +
                                     sacc[(tid + st) * 32 + d] * e2;
        }
        __syncthreads();
    }

    if (tid < 32) {
        const float val = qs[tid] + sacc[tid] / ss[0];
        const size_t off = ((size_t)(b * NQ + qi) * 128) + h * 32 + tid;
        O[off]   = val;
        Obf[off] = (_Float16)val;
    }
}

// ---------------------------------------------------------------------------
// Elementwise / prep kernels
// ---------------------------------------------------------------------------
__global__ void prep_masks(const float* __restrict__ vals, const float* __restrict__ obs,
                           const float* __restrict__ tgt,
                           float* valsf, float* obsf, float* mkf, float* tgtf,
                           int* tind, int* cind)
{
    const int i = blockIdx.x * 256 + threadIdx.x;
    if (i >= cBn) return;
    const float o = obs[i], t = tgt[i];
    const bool comb = (o > 0.0f) || (t > 0.0f);
    const float mk = comb ? 1.0f : 0.0f;
    const int n = i % cN;
    tind[i] = comb ? (n / cD) : 0;
    cind[i] = comb ? (n % cD) : 0;
    mkf[i]   = mk;
    tgtf[i]  = t * mk;
    valsf[i] = vals[i] * mk;
    obsf[i]  = o * mk;
}

__global__ void u0_kernel(const float* __restrict__ w, const float* __restrict__ bias,
                          const float* __restrict__ valsf, const float* __restrict__ obsf,
                          const float* __restrict__ mkf,
                          float* __restrict__ U, _Float16* __restrict__ Ubf)
{
    const int i = blockIdx.x * 256 + threadIdx.x;
    if (i >= cBn * 128) return;
    const int row = i >> 7, m = i & 127;
    float v = w[m * 2] * valsf[row] + w[m * 2 + 1] * obsf[row] + bias[m];
    v = fmaxf(v, 0.0f) * mkf[row];
    U[i] = v;
    Ubf[i] = (_Float16)v;
}

__global__ void time_embed(const float* __restrict__ w, const float* __restrict__ bias,
                           const float* __restrict__ tp,
                           float* __restrict__ Te, _Float16* __restrict__ Tb)
{
    const int i = blockIdx.x * 256 + threadIdx.x;
    if (i >= cB * cT * 128) return;
    const int row = i >> 7, m = i & 127;
    const float v = sinf(w[m] * tp[row] + bias[m]);
    Te[i] = v;
    Tb[i] = (_Float16)v;
}

__global__ void chan_embed(const float* __restrict__ w, const float* __restrict__ bias,
                           float* __restrict__ Ce, _Float16* __restrict__ Cb)
{
    const int i = blockIdx.x * 256 + threadIdx.x;
    if (i >= cB * cD * 128) return;
    const int bd = i >> 7, m = i & 127;
    const int d = bd % cD;
    const float v = fmaxf(w[m * cD + d] + bias[m], 0.0f);
    Ce[i] = v;
    Cb[i] = (_Float16)v;
}

__global__ void f2h_kernel(const float* __restrict__ s, _Float16* __restrict__ d, int n)
{
    const int i = blockIdx.x * 256 + threadIdx.x;
    if (i < n) d[i] = (_Float16)s[i];
}

// X[row, 0:128] = Ebf[b, ind[row], :], X[row, 128:256] = Ubf[row, :]
__global__ void build_cat2(const _Float16* __restrict__ Ebf, const _Float16* __restrict__ Ubf,
                           const int* __restrict__ ind, _Float16* __restrict__ X, int rowsE)
{
    const int i = blockIdx.x * 256 + threadIdx.x;
    if (i >= cBn * 256) return;
    const int row = i >> 8, j = i & 255;
    _Float16 v;
    if (j < 128) {
        const int b = row / cN;
        v = Ebf[((size_t)(b * rowsE + ind[row])) * 128 + j];
    } else {
        v = Ubf[(size_t)row * 128 + (j - 128)];
    }
    X[i] = v;
}

// X[row, :] = [ Tbf[b, tind, :], Cbf[b, cind, :], Ubf[row, :] ]   (K = 384)
__global__ void build_cat3(const _Float16* __restrict__ Tbf, const _Float16* __restrict__ Cbf,
                           const _Float16* __restrict__ Ubf,
                           const int* __restrict__ tind, const int* __restrict__ cind,
                           _Float16* __restrict__ X)
{
    const int i = blockIdx.x * 256 + threadIdx.x;
    if (i >= cBn * 384) return;
    const int row = i / 384;
    const int j = i - row * 384;
    const int b = row / cN;
    _Float16 v;
    if (j < 128)      v = Tbf[((size_t)(b * cT + tind[row])) * 128 + j];
    else if (j < 256) v = Cbf[((size_t)(b * cD + cind[row])) * 128 + (j - 128)];
    else              v = Ubf[(size_t)row * 128 + (j - 256)];
    X[i] = v;
}

// preds = E @ w_out^T + b; y = preds * tgt  (identity permutation -> flat order)
__global__ void out_kernel(const _Float16* __restrict__ E, const float* __restrict__ w,
                           const float* __restrict__ bsc, const float* __restrict__ tgtf,
                           float* __restrict__ y)
{
    const int i = blockIdx.x * 256 + threadIdx.x;
    if (i >= cBn) return;
    const _Float16* e = E + (size_t)i * 384;
    float acc = bsc[0];
    for (int j = 0; j < 384; ++j) acc += (float)e[j] * w[j];
    y[i] = acc * tgtf[i];
}

// ---------------------------------------------------------------------------
// Host orchestration
// ---------------------------------------------------------------------------
struct Lin { const float* w; const float* b; };
struct Mab { Lin q, k, v, o; };

extern "C" void kernel_launch(void* const* d_in, const int* in_sizes, int n_in,
                              void* d_out, int out_size, void* d_ws, size_t ws_size,
                              hipStream_t stream)
{
    (void)in_sizes; (void)n_in; (void)out_size; (void)ws_size;

    // ----- input pointer map (setup_inputs() dict order, depth-first) -----
    const float* time_points = (const float*)d_in[0];
    const float* values      = (const float*)d_in[1];
    const float* obs_mask    = (const float*)d_in[2];
    const float* target_mask = (const float*)d_in[3];
    int pi = 4;
    auto nextLin = [&]() { Lin l{(const float*)d_in[pi], (const float*)d_in[pi + 1]}; pi += 2; return l; };
    Lin edge_init = nextLin();
    Lin chan_init = nextLin();
    Lin time_init = nextLin();
    Mab ct[cL], tc[cL];
    for (int i = 0; i < cL; ++i) ct[i] = Mab{nextLin(), nextLin(), nextLin(), nextLin()};
    for (int i = 0; i < cL; ++i) tc[i] = Mab{nextLin(), nextLin(), nextLin(), nextLin()};
    Lin edgeL[cL];
    for (int i = 0; i < cL; ++i) edgeL[i] = nextLin();
    Lin outL = nextLin();

    // ----- workspace bump allocator -----
    char* ws = (char*)d_ws;
    size_t off = 0;
    auto alloc = [&](size_t bytes) -> void* {
        void* p = ws + off;
        off += (bytes + 255) & ~(size_t)255;
        return p;
    };

    float* valsf = (float*)alloc(cBn * 4);
    float* obsf  = (float*)alloc(cBn * 4);
    float* mkf   = (float*)alloc(cBn * 4);
    float* tgtf  = (float*)alloc(cBn * 4);
    int*   tind  = (int*)alloc(cBn * 4);
    int*   cind  = (int*)alloc(cBn * 4);
    float*    U   = (float*)alloc((size_t)cBn * 128 * 4);
    _Float16* Ubf = (_Float16*)alloc((size_t)cBn * 128 * 2);
    float*    Tem[2]; _Float16* Tbf[2];
    float*    Cem[2]; _Float16* Cbf[2];
    for (int j = 0; j < 2; ++j) {
        Tem[j] = (float*)alloc((size_t)cB * cT * 128 * 4);
        Tbf[j] = (_Float16*)alloc((size_t)cB * cT * 128 * 2);
        Cem[j] = (float*)alloc((size_t)cB * cD * 128 * 4);
        Cbf[j] = (_Float16*)alloc((size_t)cB * cD * 128 * 2);
    }
    _Float16* X     = (_Float16*)alloc((size_t)cBn * 384 * 2);  // shared KT / KC / E
    float*    kproj = (float*)alloc((size_t)cBn * 128 * 4);
    float*    vproj = (float*)alloc((size_t)cBn * 128 * 4);
    float*    qproj = (float*)alloc((size_t)cB * cT * 128 * 4);
    float*    Obuf  = (float*)alloc((size_t)cB * cT * 128 * 4);
    _Float16* Obf   = (_Float16*)alloc((size_t)cB * cT * 128 * 2);

    auto conv = [&](const float* src, size_t cnt) -> _Float16* {
        _Float16* dst = (_Float16*)alloc(cnt * 2);
        f2h_kernel<<<dim3((unsigned)((cnt + 255) / 256)), dim3(256), 0, stream>>>(src, dst, (int)cnt);
        return dst;
    };

    _Float16 *qh_ct[cL], *kh_ct[cL], *vh_ct[cL], *oh_ct[cL];
    _Float16 *qh_tc[cL], *kh_tc[cL], *vh_tc[cL], *oh_tc[cL];
    _Float16 *eh[cL];
    for (int i = 0; i < cL; ++i) {
        qh_ct[i] = conv(ct[i].q.w, 128 * 128);
        kh_ct[i] = conv(ct[i].k.w, 128 * 256);
        vh_ct[i] = conv(ct[i].v.w, 128 * 256);
        oh_ct[i] = conv(ct[i].o.w, 128 * 128);
        qh_tc[i] = conv(tc[i].q.w, 128 * 128);
        kh_tc[i] = conv(tc[i].k.w, 128 * 256);
        vh_tc[i] = conv(tc[i].v.w, 128 * 256);
        oh_tc[i] = conv(tc[i].o.w, 128 * 128);
        eh[i]    = conv(edgeL[i].w, 128 * 384);
    }

    // proj:  Y = X @ W^T + b                  (no relu/resid/scale, fp32 only)
    auto gemm_proj = [&](const _Float16* Xp, const _Float16* Wp, const float* bias,
                         float* Yp, int rows, int K) {
        dim3 g((rows + 15) / 16), blk(256);
        if (K == 128)
            gemm_f16_wmma<128, false, false, false, false><<<g, blk, 0, stream>>>(
                Xp, Wp, bias, nullptr, nullptr, Yp, nullptr, rows);
        else
            gemm_f16_wmma<256, false, false, false, false><<<g, blk, 0, stream>>>(
                Xp, Wp, bias, nullptr, nullptr, Yp, nullptr, rows);
    };
    // fc_o:  Y = resid + relu(X @ W^T + b)    (fp32 + f16 outputs)
    auto gemm_fc = [&](const _Float16* Xp, const _Float16* Wp, const float* bias,
                       const float* resid, float* Yp, _Float16* Ybfp, int rows) {
        dim3 g((rows + 15) / 16), blk(256);
        gemm_f16_wmma<128, true, true, false, true><<<g, blk, 0, stream>>>(
            Xp, Wp, bias, resid, nullptr, Yp, Ybfp, rows);
    };
    // edge:  Y = resid + mk * relu(X @ W^T + b)  (fp32 + f16 outputs)
    auto gemm_edge = [&](const _Float16* Xp, const _Float16* Wp, const float* bias,
                         const float* resid, const float* rs, float* Yp, _Float16* Ybfp,
                         int rows) {
        dim3 g((rows + 15) / 16), blk(256);
        gemm_f16_wmma<384, true, true, true, true><<<g, blk, 0, stream>>>(
            Xp, Wp, bias, resid, rs, Yp, Ybfp, rows);
    };

    const float scale = 1.0f / sqrtf((float)cM);

    // ----- prep -----
    prep_masks<<<dim3((cBn + 255) / 256), dim3(256), 0, stream>>>(
        values, obs_mask, target_mask, valsf, obsf, mkf, tgtf, tind, cind);
    u0_kernel<<<dim3((cBn * 128 + 255) / 256), dim3(256), 0, stream>>>(
        edge_init.w, edge_init.b, valsf, obsf, mkf, U, Ubf);
    time_embed<<<dim3((cB * cT * 128 + 255) / 256), dim3(256), 0, stream>>>(
        time_init.w, time_init.b, time_points, Tem[0], Tbf[0]);
    chan_embed<<<dim3((cB * cD * 128 + 255) / 256), dim3(256), 0, stream>>>(
        chan_init.w, chan_init.b, Cem[0], Cbf[0]);

    // ----- layers -----
    int cur = 0;
    for (int i = 0; i < cL; ++i) {
        const int nxt = cur ^ 1;

        // --- MAB ct: queries = channels, keys = [gather(T, tind), U] ---
        build_cat2<<<dim3((cBn * 256 + 255) / 256), dim3(256), 0, stream>>>(
            Tbf[cur], Ubf, tind, X, cT);
        gemm_proj(Cbf[cur], qh_ct[i], ct[i].q.b, qproj, cB * cD, 128);
        gemm_proj(X, kh_ct[i], ct[i].k.b, kproj, cBn, 256);
        gemm_proj(X, vh_ct[i], ct[i].v.b, vproj, cBn, 256);
        attn_kernel<<<dim3(cD, cH, cB), dim3(256), 0, stream>>>(
            qproj, kproj, vproj, cind, mkf, Obuf, Obf, cD, cN, scale);
        gemm_fc(Obf, oh_ct[i], ct[i].o.b, Obuf, Cem[nxt], Cbf[nxt], cB * cD);

        // --- MAB tc: queries = times, keys = [gather(C_old, cind), U] ---
        build_cat2<<<dim3((cBn * 256 + 255) / 256), dim3(256), 0, stream>>>(
            Cbf[cur], Ubf, cind, X, cD);
        gemm_proj(Tbf[cur], qh_tc[i], tc[i].q.b, qproj, cB * cT, 128);
        gemm_proj(X, kh_tc[i], tc[i].k.b, kproj, cBn, 256);
        gemm_proj(X, vh_tc[i], tc[i].v.b, vproj, cBn, 256);
        attn_kernel<<<dim3(cT, cH, cB), dim3(256), 0, stream>>>(
            qproj, kproj, vproj, tind, mkf, Obuf, Obf, cT, cN, scale);
        gemm_fc(Obf, oh_tc[i], tc[i].o.b, Obuf, Tem[nxt], Tbf[nxt], cB * cT);

        // --- edge update: U = relu(W_e [T_new|C_new|U] + b) * mk + U ---
        build_cat3<<<dim3((cBn * 384 + 255) / 256), dim3(256), 0, stream>>>(
            Tbf[nxt], Cbf[nxt], Ubf, tind, cind, X);
        gemm_edge(X, eh[i], edgeL[i].b, U, mkf, U, Ubf, cBn);

        cur = nxt;
    }

    // ----- output head -----
    build_cat3<<<dim3((cBn * 384 + 255) / 256), dim3(256), 0, stream>>>(
        Tbf[cur], Cbf[cur], Ubf, tind, cind, X);
    out_kernel<<<dim3((cBn + 255) / 256), dim3(256), 0, stream>>>(
        X, outL.w, outL.b, tgtf, (float*)d_out);
}
